// MechanismNet_40879498729286
// MI455X (gfx1250) — matrix operand order repaired and checked by reading, not compiled, not run on hardware
//
#include <hip/hip_runtime.h>
#include <hip/hip_bf16.h>

// ---------------------------------------------------------------------------
// MechanismNet forward on MI455X (gfx1250, wave32):
//   one-hot gather-sum (L1, W1 is L2-resident: 128MB < 192MB L2)
//   -> bf16 WMMA GEMM with async-to-LDS B staging (L2, relu)
//   -> bf16 WMMA GEMM (L3, f32 logits)
//   -> potential-form log-space Sinkhorn (80 iters, wave-per-matrix, f32)
// ---------------------------------------------------------------------------

typedef __attribute__((ext_vector_type(16))) __bf16 v16bf;
typedef __attribute__((ext_vector_type(8)))  float  v8f;

union FragBF {
    uint4 u[2];
    v16bf v;
};

#define HID     1024
#define BATCH   2048
#define ITERS   80
#define SEPS    1e-12f

// ---------------------------------------------------------------------------
// Weight transpose + f32 -> bf16 convert:  Wt[n][k] = (bf16) W[k][n]
// ---------------------------------------------------------------------------
__global__ __launch_bounds__(256) void wtrans_kernel(
    const float* __restrict__ W, __hip_bfloat16* __restrict__ Wt)
{
    __shared__ float tile[32][33];
    const int tx = threadIdx.x;
    const int k0 = blockIdx.y * 32;
    const int n0 = blockIdx.x * 32;
    for (int r = threadIdx.y; r < 32; r += 8)
        tile[r][tx] = W[(size_t)(k0 + r) * HID + (n0 + tx)];
    __syncthreads();
    for (int r = threadIdx.y; r < 32; r += 8)
        Wt[(size_t)(n0 + r) * HID + (k0 + tx)] = __float2bfloat16(tile[tx][r]);
}

// ---------------------------------------------------------------------------
// Layer 1 gather-sum: h1[b,:] = relu(b1 + sum_k W1[k*32 + profile[b,k], :])
// One block per batch item; each gathered 4KB W1 row is one fully-coalesced
// 256 x float4 read.  grid 2048, block 256.
// ---------------------------------------------------------------------------
__global__ __launch_bounds__(256) void embed_layer1_kernel(
    const long long* __restrict__ profile, const float* __restrict__ W1,
    const float* __restrict__ b1, __hip_bfloat16* __restrict__ h1)
{
    __shared__ int sIdx[1024];
    const int b = blockIdx.x;
    const int t = threadIdx.x;

    for (int k = t; k < 1024; k += 256) {
        const int p = (int)profile[(size_t)b * 1024 + k];
        sIdx[k] = (k << 5) + p;
    }
    __syncthreads();

    const int c0 = t * 4;
    float4 acc = *(const float4*)(b1 + c0);
    #pragma unroll 4
    for (int k = 0; k < 1024; ++k) {
        const float4 v = *(const float4*)(W1 + (size_t)sIdx[k] * HID + c0);
        acc.x += v.x; acc.y += v.y; acc.z += v.z; acc.w += v.w;
    }
    const size_t o = (size_t)b * HID + c0;
    h1[o + 0] = __float2bfloat16(fmaxf(acc.x, 0.0f));
    h1[o + 1] = __float2bfloat16(fmaxf(acc.y, 0.0f));
    h1[o + 2] = __float2bfloat16(fmaxf(acc.z, 0.0f));
    h1[o + 3] = __float2bfloat16(fmaxf(acc.w, 0.0f));
}

// ---------------------------------------------------------------------------
// bf16 WMMA GEMM, out[2048 x 1024] = A @ Bt^T + bias.
// Block = 8 waves = 128(M) x 64(N) tile; grid 256 (16 M-blocks x 16 N-blocks).
// K is consumed in chunks of 64 (two 32-wide WMMA slabs per chunk):
//   * B slab 64(N) x 64(K) bf16 (8KB) staged into double-buffered LDS via
//     2x global_load_async_to_lds_b128 per thread (ASYNCcnt-tracked,
//     s_wait_asynccnt 2 pipelines the next chunk behind current compute);
//     8x reuse across the block's waves, 2 barriers per 8 wmma.
//   * A fragments direct from global (A 16x32 bf16 layout: lane l ->
//     row l%16, K {c..c+7, c+16..c+23}, c = 8*(l>=16)).
//   * B fragments pre-loaded from LDS as ds_load_b128 pairs (B 32x16 bf16
//     layout: lane l -> col l%16, K {c..c+7, c+16..c+23}, c = 8*(l>=16)), then the 4 wmma
//     of a half-step issue back-to-back (no per-wmma dscnt stalls).
// Outer K loop is unroll(1) so accumulators stay pinned in one VGPR set
// (avoids the unroll-by-2 acc rotation + WMMA->VALU hazard nops seen in r2).
// C/D f32 16x16: lane l -> N = l%16, VGPR r -> M = r + 8*(l>=16).
// MODE 0: relu -> bf16 ; MODE 1: f32 logits.
// ---------------------------------------------------------------------------
template <int MODE>
__global__ __launch_bounds__(256) void gemm_wmma_kernel(
    const __hip_bfloat16* __restrict__ A,
    const __hip_bfloat16* __restrict__ Bt,
    const float* __restrict__ bias,
    __hip_bfloat16* __restrict__ outb,
    float* __restrict__ outf)
{
    constexpr int K = HID;
    __shared__ __align__(16) __hip_bfloat16 Bs[2][64][64];   // 2 x 8KB

    const int mb   = blockIdx.x & 15;      // M block: rows [mb*128, +128)
    const int nb   = blockIdx.x >> 4;      // N block: cols [nb*64, +64)
    const int w    = threadIdx.x >> 5;
    const int lane = threadIdx.x & 31;
    const int lm   = lane & 15;
    const int hi   = lane >> 4;
    const int t    = threadIdx.x;

    const int nBase = nb * 64;
    const int mTile = mb * 8 + w;

    // B staging map: thread t -> row n = t/4, chunks (t%4)*8 and +32
    const int sn = t >> 2;
    const int sc = (t & 3) * 8;
    const __hip_bfloat16* gB = Bt + (size_t)(nBase + sn) * K + sc;

    const size_t aOff = (size_t)(mTile * 16 + lm) * K + (size_t)(hi * 8);

    v8f acc[4];
    #pragma unroll
    for (int s = 0; s < 4; ++s) {
        const float bv = bias[nBase + s * 16 + lm];
        #pragma unroll
        for (int r = 0; r < 8; ++r) acc[s][r] = bv;
    }

    auto issue = [&](int kk) {        // kk = 64-wide chunk index, 0..15
        const __hip_bfloat16* g0 = gB + kk * 64;
        const unsigned l0 = (unsigned)(size_t)(&Bs[kk & 1][sn][sc]);
        asm volatile("global_load_async_to_lds_b128 %0, %1, off"
                     :: "v"(l0), "v"(g0) : "memory");
        const __hip_bfloat16* g1 = g0 + 32;
        const unsigned l1 = l0 + 64;  // +32 bf16 elements
        asm volatile("global_load_async_to_lds_b128 %0, %1, off"
                     :: "v"(l1), "v"(g1) : "memory");
    };

    issue(0);
    #pragma unroll 1
    for (int kk = 0; kk < 16; ++kk) {
        if (kk + 1 < 16) {
            issue(kk + 1);
            asm volatile("s_wait_asynccnt 0x2" ::: "memory");
        } else {
            asm volatile("s_wait_asynccnt 0x0" ::: "memory");
        }
        __syncthreads();   // chunk kk visible to all waves

        const __hip_bfloat16(*buf)[64] = Bs[kk & 1];
        #pragma unroll
        for (int h2 = 0; h2 < 2; ++h2) {          // two 32-wide K slabs
            const int k0 = kk * 64 + h2 * 32;
            FragBF a;
            a.u[0] = *(const uint4*)(A + aOff + k0);
            a.u[1] = *(const uint4*)(A + aOff + k0 + 16);

            FragBF bfg[4];
            #pragma unroll
            for (int s = 0; s < 4; ++s) {
                const __hip_bfloat16* bp = &buf[s * 16 + lm][h2 * 32 + hi * 8];
                bfg[s].u[0] = *(const uint4*)(bp);
                bfg[s].u[1] = *(const uint4*)(bp + 16);
            }
            #pragma unroll
            for (int s = 0; s < 4; ++s)
                acc[s] = __builtin_amdgcn_wmma_f32_16x16x32_bf16(
                    false, a.v, false, bfg[s].v, (short)0, acc[s], false, false);
        }
        __syncthreads();   // all waves done with chunk kk before it is reused
    }

    const int mOut = mTile * 16 + hi * 8;
    #pragma unroll
    for (int s = 0; s < 4; ++s) {
        const int n = nBase + s * 16 + lm;
        #pragma unroll
        for (int r = 0; r < 8; ++r) {
            const float v = acc[s][r];
            const size_t o = (size_t)(mOut + r) * HID + n;
            if (MODE == 0) outb[o] = __float2bfloat16(v > 0.0f ? v : 0.0f);
            else           outf[o] = v;
        }
    }
}

// ---------------------------------------------------------------------------
// Sinkhorn in potential form, one wave per 32x32 matrix (wave32-native).
// lk_t = K0 + r_i + c_j with per-iteration updates
//   r_i = -lse_j(K0[i,j] + c_j) ;  c_j = -lse_i(K0[i,j] + r_i)
// (exactly equivalent to the reference's in-place row/col normalization).
// Lane i holds row i and column i of K0 in registers (64 VGPRs); each pass is
// 8 broadcast ds_load_b128 of the 32-float potential vector + ONE ds_store.
// All LDS data is wave-private and same-wave LDS ops are in-order, so only
// compiler scheduling fences (wave_barrier) are needed: zero s_barriers in
// the hot loop.  grid 256, block 256 (8 waves); ~35.8 KB LDS of 320 KB/WGP.
// ---------------------------------------------------------------------------
__global__ __launch_bounds__(256) void sinkhorn_kernel(
    const float* __restrict__ logits, float* __restrict__ out)
{
    struct WaveLds {
        float Mm[32 * 33];                       // 4224 B (16B-aligned size)
        __align__(16) float rv[32];
        __align__(16) float cv[32];
    };
    __shared__ WaveLds lds[8];
    const int w    = threadIdx.x >> 5;
    const int lane = threadIdx.x & 31;
    WaveLds& L = lds[w];
    const size_t base = ((size_t)(blockIdx.x << 3) + w) * 1024;

    // coalesced staged load + nan_to_num(nan=0, +/-inf=0)
    #pragma unroll 4
    for (int i = 0; i < 32; ++i) {
        float v = logits[base + i * 32 + lane];
        const unsigned a = __float_as_uint(v) & 0x7fffffffu;
        if (a >= 0x7f800000u) v = 0.0f;
        L.Mm[i * 33 + lane] = v;          // pitch 33: conflict-free rows+cols
    }
    __builtin_amdgcn_wave_barrier();

    float K0r[32], K0c[32];
    #pragma unroll
    for (int j = 0; j < 32; ++j) K0r[j] = L.Mm[lane * 33 + j];  // row  lane
    #pragma unroll
    for (int j = 0; j < 32; ++j) K0c[j] = L.Mm[j * 33 + lane];  // col  lane

    L.cv[lane] = 0.0f;
    __builtin_amdgcn_wave_barrier();

    const float4* cv4 = (const float4*)L.cv;
    const float4* rv4 = (const float4*)L.rv;

    float vv[32];
    for (int it = 0; it < ITERS; ++it) {
        // ---- r_i = -lse_j(K0[i,j] + c_j) ----
        float mx = -3.4e38f;
        #pragma unroll
        for (int q = 0; q < 8; ++q) {             // 8x ds_load_b128 broadcast
            const float4 cc = cv4[q];
            vv[4 * q + 0] = K0r[4 * q + 0] + cc.x;
            vv[4 * q + 1] = K0r[4 * q + 1] + cc.y;
            vv[4 * q + 2] = K0r[4 * q + 2] + cc.z;
            vv[4 * q + 3] = K0r[4 * q + 3] + cc.w;
            mx = fmaxf(mx, fmaxf(fmaxf(vv[4 * q + 0], vv[4 * q + 1]),
                                 fmaxf(vv[4 * q + 2], vv[4 * q + 3])));
        }
        float s = 0.0f;
        #pragma unroll
        for (int j = 0; j < 32; ++j) s += __expf(vv[j] - mx);
        float r = -(mx + __logf(s));
        if (r != r) r = 0.0f;                       // nan guard
        r = fminf(fmaxf(r, -1.0e30f), 1.0e30f);     // inf guard
        L.rv[lane] = r;
        __builtin_amdgcn_wave_barrier();

        // ---- c_j = -lse_i(K0[i,j] + r_i) ----
        mx = -3.4e38f;
        #pragma unroll
        for (int q = 0; q < 8; ++q) {
            const float4 rr = rv4[q];
            vv[4 * q + 0] = K0c[4 * q + 0] + rr.x;
            vv[4 * q + 1] = K0c[4 * q + 1] + rr.y;
            vv[4 * q + 2] = K0c[4 * q + 2] + rr.z;
            vv[4 * q + 3] = K0c[4 * q + 3] + rr.w;
            mx = fmaxf(mx, fmaxf(fmaxf(vv[4 * q + 0], vv[4 * q + 1]),
                                 fmaxf(vv[4 * q + 2], vv[4 * q + 3])));
        }
        s = 0.0f;
        #pragma unroll
        for (int i = 0; i < 32; ++i) s += __expf(vv[i] - mx);
        float c = -(mx + __logf(s));
        if (c != c) c = 0.0f;
        c = fminf(fmaxf(c, -1.0e30f), 1.0e30f);
        L.cv[lane] = c;
        __builtin_amdgcn_wave_barrier();
    }

    // ---- finalize: x = max(exp(K0+r+c), eps); row-norm; col-norm; clamp ----
    {
        const float r = L.rv[lane];
        float s = 0.0f;
        #pragma unroll
        for (int q = 0; q < 8; ++q) {
            const float4 cc = cv4[q];
            const float x0 = fmaxf(__expf(K0r[4 * q + 0] + r + cc.x), SEPS);
            const float x1 = fmaxf(__expf(K0r[4 * q + 1] + r + cc.y), SEPS);
            const float x2 = fmaxf(__expf(K0r[4 * q + 2] + r + cc.z), SEPS);
            const float x3 = fmaxf(__expf(K0r[4 * q + 3] + r + cc.w), SEPS);
            vv[4 * q + 0] = x0; vv[4 * q + 1] = x1;
            vv[4 * q + 2] = x2; vv[4 * q + 3] = x3;
            s += (x0 + x1) + (x2 + x3);
        }
        const float invr = 1.0f / (s + SEPS);
        #pragma unroll
        for (int j = 0; j < 32; ++j) L.Mm[lane * 33 + j] = vv[j] * invr;
        __builtin_amdgcn_wave_barrier();

        s = 0.0f;
        #pragma unroll
        for (int j = 0; j < 32; ++j) { vv[j] = L.Mm[j * 33 + lane]; s += vv[j]; }
        const float invc = 1.0f / (s + SEPS);
        #pragma unroll
        for (int j = 0; j < 32; ++j)
            out[base + j * 32 + lane] = fmaxf(vv[j] * invc, SEPS);  // coalesced
    }
}

// ---------------------------------------------------------------------------
// Host launcher. Inputs: profile(i64), W1, b1, W2, b2, W3, b3 (f32).
// Workspace (20 MB): [0,2M) W2t | [2M,4M) W3t | [4M,8M) h1 | [8M,12M) h2 |
//                    [12M,20M) logits f32
// ---------------------------------------------------------------------------
extern "C" void kernel_launch(void* const* d_in, const int* in_sizes, int n_in,
                              void* d_out, int out_size, void* d_ws, size_t ws_size,
                              hipStream_t stream)
{
    (void)in_sizes; (void)n_in; (void)out_size; (void)ws_size;

    const long long* profile = (const long long*)d_in[0];
    const float* W1 = (const float*)d_in[1];
    const float* b1 = (const float*)d_in[2];
    const float* W2 = (const float*)d_in[3];
    const float* b2 = (const float*)d_in[4];
    const float* W3 = (const float*)d_in[5];
    const float* b3 = (const float*)d_in[6];
    float* out = (float*)d_out;

    char* ws = (char*)d_ws;
    __hip_bfloat16* W2t = (__hip_bfloat16*)(ws);
    __hip_bfloat16* W3t = (__hip_bfloat16*)(ws + (size_t)(2)  * 1024 * 1024);
    __hip_bfloat16* h1  = (__hip_bfloat16*)(ws + (size_t)(4)  * 1024 * 1024);
    __hip_bfloat16* h2  = (__hip_bfloat16*)(ws + (size_t)(8)  * 1024 * 1024);
    float* logits       = (float*)         (ws + (size_t)(12) * 1024 * 1024);

    const dim3 tg(32, 32), tb(32, 8);
    wtrans_kernel<<<tg, tb, 0, stream>>>(W2, W2t);
    wtrans_kernel<<<tg, tb, 0, stream>>>(W3, W3t);

    embed_layer1_kernel<<<BATCH, 256, 0, stream>>>(profile, W1, b1, h1);

    gemm_wmma_kernel<0><<<256, 256, 0, stream>>>(h1, W2t, b2, h2, nullptr);
    gemm_wmma_kernel<1><<<256, 256, 0, stream>>>(h2, W3t, b3, nullptr, logits);

    sinkhorn_kernel<<<256, 256, 0, stream>>>(logits, out);
}
